// GRU_88407606821524
// MI455X (gfx1250) — matrix operand order repaired and checked by reading, not compiled, not run on hardware
//
#include <hip/hip_runtime.h>
#include <hip/hip_bf16.h>

// ---------------------------------------------------------------------------
// Problem constants (match reference)
// ---------------------------------------------------------------------------
#define BB   512          // batch
#define SS   512          // sequence length
#define CC   32           // input channels
#define HH   512          // hidden
#define OUTN 3072         // P*C = 96*32
#define LDH  528          // LDS row stride in bf16 (H + 16 pad, keeps 16B align)
#define NTHREADS 512      // 16 waves -> 4 waves per SIMD32
#define NWAVES   16
#define TPW  2            // gate N-tiles per wave (32 tiles / 16 waves)
#define PPW  12           // projection N-tiles per wave (192 / 16)
#define KT   16           // K tiles over H (512/32)

typedef __attribute__((ext_vector_type(16))) __bf16 v16bf;
typedef __attribute__((ext_vector_type(8)))  float  v8f;

// ---------------------------------------------------------------------------
// fp32 -> bf16 (round to nearest even), bit-exact with packed layout
// ---------------------------------------------------------------------------
__device__ __forceinline__ unsigned int f2bf_bits(float f) {
    union { float f; unsigned int u; } a; a.f = f;
    unsigned int u = a.u;
    u += 0x7FFFu + ((u >> 16) & 1u);
    return u >> 16;
}

// ---------------------------------------------------------------------------
// Weight packing: fp32 row-major W[N][K] -> bf16 WMMA B-fragments.
// Tile (ntile,ktile) -> 256 dwords; dword (lane,j):
//   col n = ntile*16 + lane%16
//   k     = ktile*32 + (lane/16)*8 + 2*j + (j/4)*8   (pair k, k+1 packed lo/hi)
// matches A layout: lanes 0-15 hold K {0..7,16..23}, lanes 16-31 hold K {8..15,24..31}.
// ---------------------------------------------------------------------------
__global__ void pack_bf16_b(const float* __restrict__ W, unsigned int* __restrict__ dst,
                            int K, int Ktiles, int total) {
    int idx = blockIdx.x * blockDim.x + threadIdx.x;
    if (idx >= total) return;
    int tile  = idx >> 8;
    int r     = idx & 255;
    int lane  = r >> 3;
    int j     = r & 7;
    int ntile = tile / Ktiles;
    int ktile = tile - ntile * Ktiles;
    int n = ntile * 16 + (lane & 15);
    int k = ktile * 32 + ((lane >> 4) << 3) + (j << 1) + ((j >> 2) << 3);
    const float* p = W + (size_t)n * K + k;
    dst[idx] = f2bf_bits(p[0]) | (f2bf_bits(p[1]) << 16);
}

// ---------------------------------------------------------------------------
// Fragment helpers
// ---------------------------------------------------------------------------
__device__ __forceinline__ v8f wmma_bf16(v16bf a, v16bf b, v8f c) {
    return __builtin_amdgcn_wmma_f32_16x16x32_bf16(false, a, false, b, (short)0, c, false, false);
}

// B fragment from packed global/L2 buffer: 2x global_load_b128 per lane
__device__ __forceinline__ v16bf bfrag(const unsigned int* __restrict__ pk, int tile, int lane) {
    const unsigned int* p = pk + ((size_t)tile << 8) + (lane << 3);
    union { v16bf v; uint4 q[2]; } u;
    u.q[0] = *(const uint4*)(p);
    u.q[1] = *(const uint4*)(p + 4);
    return u.v;
}

// Prefetch one packed B tile (1 KB) into cache: 32 lanes x 32B stride covers it.
__device__ __forceinline__ void bprefetch(const unsigned int* __restrict__ pk, int tile, int lane) {
    __builtin_prefetch(pk + ((size_t)tile << 8) + (lane << 3), 0, 1);
}

// A fragment from LDS bf16 matrix [16][LDH], row-major: 2x ds_load_b128 per lane.
// 16-bit A layout: lanes 0-15 row m, K {0..7,16..23}; lanes 16-31 row m, K {8..15,24..31}.
__device__ __forceinline__ v16bf lds_afrag(const __bf16* base, int lane, int kbase) {
    int row = lane & 15;
    int k0  = kbase + ((lane & 16) ? 8 : 0);
    const __bf16* p = base + row * LDH + k0;
    union { v16bf v; uint4 q[2]; } u;
    u.q[0] = *(const uint4*)(p);
    u.q[1] = *(const uint4*)(p + 16);
    return u.v;
}

// A fragment from global fp32 x (B,S,C), K = C = 32, converted to bf16 on the fly
__device__ __forceinline__ v16bf x_afrag(const float* __restrict__ x, int bbase, int t, int lane) {
    int row = lane & 15;
    int k0  = (lane & 16) ? 8 : 0;
    const float* p = x + ((size_t)(bbase + row) * SS + t) * CC + k0;
    union { v16bf v; __bf16 h[16]; } u;
#pragma unroll
    for (int j = 0; j < 8; ++j) {
        u.h[j]     = (__bf16)p[j];        // K = k0 .. k0+7
        u.h[8 + j] = (__bf16)p[16 + j];   // K = k0+16 .. k0+23
    }
    return u.v;
}

__device__ __forceinline__ float sigmoidf(float v) { return 1.f / (1.f + __expf(-v)); }

// ---------------------------------------------------------------------------
// Persistent GRU kernel: 1 workgroup = 16 batch rows, entire sequence.
// 512 threads = 16 waves (4 per SIMD); each wave owns 2 of the 32 N-tiles/gate.
// ---------------------------------------------------------------------------
__global__ __launch_bounds__(NTHREADS)
void gru_persist(const float* __restrict__ x,
                 const unsigned int* __restrict__ pWi0,
                 const unsigned int* __restrict__ pWh0,
                 const float* __restrict__ b0,
                 const unsigned int* __restrict__ pWi1,
                 const unsigned int* __restrict__ pWh1,
                 const float* __restrict__ b1,
                 const unsigned int* __restrict__ pWp,
                 const float* __restrict__ bp,
                 float* __restrict__ out) {
    __shared__ __bf16 sh0[16 * LDH];   // layer-0 hidden state (bf16)
    __shared__ __bf16 sh1[16 * LDH];   // layer-1 hidden state
    __shared__ __bf16 srh[16 * LDH];   // r * h staging buffer

    const int lane  = threadIdx.x & 31;
    const int wave  = threadIdx.x >> 5;
    const int bbase = blockIdx.x * 16;
    const int col0  = lane & 15;
    const int rowhi = (lane >> 4) << 3;   // C/D frag: lanes 16-31 hold rows 8..15

    for (int i = threadIdx.x; i < 16 * LDH; i += NTHREADS) {
        sh0[i] = (__bf16)0.f;
        sh1[i] = (__bf16)0.f;
    }
    __syncthreads();

    for (int t = 0; t < SS; ++t) {
        v16bf ax = x_afrag(x, bbase, t, lane);
        float zf[TPW][8];

        // ---------------- layer 0, phase A: r, z ----------------
#pragma unroll
        for (int i = 0; i < TPW; ++i) {
            int nt = wave * TPW + i;
            // warm the phase-B n-gate weight stream across the barrier
            bprefetch(pWh0, (64 + nt) * KT, lane);
            v8f ar = {}; v8f az = {};
            ar = wmma_bf16(ax, bfrag(pWi0, nt, lane), ar);             // K = C = 32, single tile
            az = wmma_bf16(ax, bfrag(pWi0, 32 + nt, lane), az);
#pragma unroll 4
            for (int kt = 0; kt < KT; ++kt) {
                v16bf ah = lds_afrag(sh0, lane, kt * 32);
                ar = wmma_bf16(ah, bfrag(pWh0, nt * KT + kt, lane), ar);
                az = wmma_bf16(ah, bfrag(pWh0, (32 + nt) * KT + kt, lane), az);
            }
            int col = nt * 16 + col0;
            float br = b0[col], bz = b0[HH + col];
#pragma unroll
            for (int v = 0; v < 8; ++v) {
                float r  = sigmoidf(ar[v] + br);
                zf[i][v] = sigmoidf(az[v] + bz);
                int row  = rowhi + v;
                float hv = (float)sh0[row * LDH + col];
                srh[row * LDH + col] = (__bf16)(r * hv);
            }
        }
        __syncthreads();

        // ---------------- layer 0, phase B: n, h0 update ----------------
#pragma unroll
        for (int i = 0; i < TPW; ++i) {
            int nt = wave * TPW + i;
            // warm the layer-1 r-gate weight streams across the barrier
            bprefetch(pWi1, nt * KT, lane);
            bprefetch(pWh1, nt * KT, lane);
            v8f an = {};
            an = wmma_bf16(ax, bfrag(pWi0, 64 + nt, lane), an);
#pragma unroll 4
            for (int kt = 0; kt < KT; ++kt) {
                v16bf arh = lds_afrag(srh, lane, kt * 32);
                an = wmma_bf16(arh, bfrag(pWh0, (64 + nt) * KT + kt, lane), an);
            }
            int col = nt * 16 + col0;
            float bn = b0[2 * HH + col];
#pragma unroll
            for (int v = 0; v < 8; ++v) {
                float n  = tanhf(an[v] + bn);
                float z  = zf[i][v];
                int row  = rowhi + v;
                float ho = (float)sh0[row * LDH + col];
                sh0[row * LDH + col] = (__bf16)((1.f - z) * n + z * ho);
            }
        }
        __syncthreads();

        // ---------------- layer 1, phase A: r, z (input = new h0) ----------------
#pragma unroll
        for (int i = 0; i < TPW; ++i) {
            int nt = wave * TPW + i;
            // warm the layer-1 n-gate weight streams across the barrier
            bprefetch(pWi1, (64 + nt) * KT, lane);
            bprefetch(pWh1, (64 + nt) * KT, lane);
            v8f ar = {}; v8f az = {};
#pragma unroll 4
            for (int kt = 0; kt < KT; ++kt) {
                v16bf a0 = lds_afrag(sh0, lane, kt * 32);
                ar = wmma_bf16(a0, bfrag(pWi1, nt * KT + kt, lane), ar);
                az = wmma_bf16(a0, bfrag(pWi1, (32 + nt) * KT + kt, lane), az);
                v16bf a1 = lds_afrag(sh1, lane, kt * 32);
                ar = wmma_bf16(a1, bfrag(pWh1, nt * KT + kt, lane), ar);
                az = wmma_bf16(a1, bfrag(pWh1, (32 + nt) * KT + kt, lane), az);
            }
            int col = nt * 16 + col0;
            float br = b1[col], bz = b1[HH + col];
#pragma unroll
            for (int v = 0; v < 8; ++v) {
                float r  = sigmoidf(ar[v] + br);
                zf[i][v] = sigmoidf(az[v] + bz);
                int row  = rowhi + v;
                float hv = (float)sh1[row * LDH + col];
                srh[row * LDH + col] = (__bf16)(r * hv);
            }
        }
        __syncthreads();

        // ---------------- layer 1, phase B: n, h1 update ----------------
#pragma unroll
        for (int i = 0; i < TPW; ++i) {
            int nt = wave * TPW + i;
            v8f an = {};
#pragma unroll 4
            for (int kt = 0; kt < KT; ++kt) {
                v16bf a0  = lds_afrag(sh0, lane, kt * 32);
                an = wmma_bf16(a0, bfrag(pWi1, (64 + nt) * KT + kt, lane), an);
                v16bf arh = lds_afrag(srh, lane, kt * 32);
                an = wmma_bf16(arh, bfrag(pWh1, (64 + nt) * KT + kt, lane), an);
            }
            int col = nt * 16 + col0;
            float bn = b1[2 * HH + col];
#pragma unroll
            for (int v = 0; v < 8; ++v) {
                float n  = tanhf(an[v] + bn);
                float z  = zf[i][v];
                int row  = rowhi + v;
                float ho = (float)sh1[row * LDH + col];
                sh1[row * LDH + col] = (__bf16)((1.f - z) * n + z * ho);
            }
        }
        __syncthreads();
    }

    // ---------------- projection: out = h1 @ Wp.T + bp ----------------
    for (int i = 0; i < PPW; ++i) {           // 192 N-tiles / 16 waves
        int nt = wave * PPW + i;
        v8f acc = {};
#pragma unroll 4
        for (int kt = 0; kt < KT; ++kt) {
            acc = wmma_bf16(lds_afrag(sh1, lane, kt * 32),
                            bfrag(pWp, nt * KT + kt, lane), acc);
        }
        int col = nt * 16 + col0;
        float bias = bp[col];
#pragma unroll
        for (int v = 0; v < 8; ++v) {
            int row = rowhi + v;
            out[(size_t)(bbase + row) * OUTN + col] = acc[v] + bias;
        }
    }
}

// ---------------------------------------------------------------------------
// Launch: pack weights into d_ws (bf16 B-fragments), then persistent GRU.
// d_ws usage: 1,990,656 dwords (~8 MB).
// ---------------------------------------------------------------------------
extern "C" void kernel_launch(void* const* d_in, const int* in_sizes, int n_in,
                              void* d_out, int out_size, void* d_ws, size_t ws_size,
                              hipStream_t stream) {
    const float* x   = (const float*)d_in[0];
    const float* Wi0 = (const float*)d_in[1];
    const float* Wh0 = (const float*)d_in[2];
    const float* b0  = (const float*)d_in[3];
    const float* Wi1 = (const float*)d_in[4];
    const float* Wh1 = (const float*)d_in[5];
    const float* b1  = (const float*)d_in[6];
    const float* Wp  = (const float*)d_in[7];
    const float* bp  = (const float*)d_in[8];

    unsigned int* ws   = (unsigned int*)d_ws;
    const int szWi0 = 96 * 1 * 256;        // 1536/16 ntiles, C=32 -> 1 ktile
    const int szWh  = 96 * 16 * 256;       // 1536/16 ntiles, 512/32 ktiles
    const int szWp  = 192 * 16 * 256;      // 3072/16 ntiles
    unsigned int* pWi0 = ws;
    unsigned int* pWh0 = pWi0 + szWi0;
    unsigned int* pWi1 = pWh0 + szWh;
    unsigned int* pWh1 = pWi1 + szWh;
    unsigned int* pWp  = pWh1 + szWh;

    pack_bf16_b<<<(szWi0 + 255) / 256, 256, 0, stream>>>(Wi0, pWi0, CC, 1, szWi0);
    pack_bf16_b<<<(szWh + 255) / 256, 256, 0, stream>>>(Wh0, pWh0, HH, KT, szWh);
    pack_bf16_b<<<(szWh + 255) / 256, 256, 0, stream>>>(Wi1, pWi1, HH, KT, szWh);
    pack_bf16_b<<<(szWh + 255) / 256, 256, 0, stream>>>(Wh1, pWh1, HH, KT, szWh);
    pack_bf16_b<<<(szWp + 255) / 256, 256, 0, stream>>>(Wp, pWp, HH, KT, szWp);

    gru_persist<<<BB / 16, NTHREADS, 0, stream>>>(x, pWi0, pWh0, b0, pWi1, pWh1, b1,
                                                  pWp, bp, (float*)d_out);
}
